// MambaEncoderFlip_68564857913690
// MI455X (gfx1250) — matrix operand order repaired and checked by reading, not compiled, not run on hardware
//
#include <hip/hip_runtime.h>
#include <hip/hip_bf16.h>
#include <math.h>

// MambaEncoderFlip for gfx1250 (MI455X, wave32, WMMA).
// Dense projections: v_wmma_f32_16x16x32_f16, weights pre-packed into WMMA
// fragment order so each lane's B-fragment is 2x global_load_b128.
// Tile count per wave is a template constant -> no spills in the k-loop.
// Selective scan: register-resident VALU kernel (1 lane per (b,d), 16 states).

typedef __attribute__((ext_vector_type(16))) _Float16 v16h;
typedef __attribute__((ext_vector_type(8)))  _Float16 v8h;
typedef __attribute__((ext_vector_type(8)))  float    v8f;

#define SEQ     1024
#define NBATCH  4
#define TOKENS  (SEQ * NBATCH)   // 4096
#define DMODEL  256
#define DINNER  512
#define DSTATE  16
#define DTRANK  16
#define INDIM   128

__device__ __forceinline__ float silu_f(float v) { return v / (1.0f + __expf(-v)); }

// ---------------------------------------------------------------------------
// Pack weight W (N x K fp32 row-major, i.e. original O x I) into WMMA B-frag
// order: Wp[nt][kt][lane][e] fp16, where
//   k = kt*32 + (lane>>4)*8 + (e&7) + (e>>3)*16,  n = nt*16 + (lane&15)
// Each lane's 16 halves are contiguous (32B) -> vector loads in the GEMM.
// ---------------------------------------------------------------------------
__global__ void k_pack_w(const float* __restrict__ W, _Float16* __restrict__ Wp,
                         int N, int K) {
    int idx = blockIdx.x * blockDim.x + threadIdx.x;
    int Ktiles = K >> 5;
    int total = (N >> 4) * Ktiles * 512;
    if (idx >= total) return;
    int e    = idx & 15;
    int lane = (idx >> 4) & 31;
    int kt   = (idx >> 9) % Ktiles;
    int nt   = (idx >> 9) / Ktiles;
    int k = kt * 32 + (lane >> 4) * 8 + (e & 7) + (e >> 3) * 16;
    int n = nt * 16 + (lane & 15);
    Wp[idx] = (_Float16)W[(size_t)n * K + k];
}

// ---------------------------------------------------------------------------
// Top RMSNorm: x (tok x D fp32) -> xn16 (tok x D fp16).  One block per token.
// ---------------------------------------------------------------------------
__global__ void k_rmsnorm16(const float* __restrict__ x, const float* __restrict__ w,
                            _Float16* __restrict__ out, int D) {
    __shared__ float red[256];
    int tok = blockIdx.x, tid = threadIdx.x;
    float v = x[(size_t)tok * D + tid];
    red[tid] = v * v;
    __syncthreads();
    for (int s = blockDim.x >> 1; s > 0; s >>= 1) {
        if (tid < s) red[tid] += red[tid + s];
        __syncthreads();
    }
    float rs = rsqrtf(red[0] / (float)D + 1e-5f);
    out[(size_t)tok * D + tid] = (_Float16)(v * rs * w[tid]);
}

// Post-norm (Mamba): out32 = rmsnorm(mo)*w + pre16  (residual).
__global__ void k_postnorm(const float* __restrict__ mo, const float* __restrict__ w,
                           const _Float16* __restrict__ pre, float* __restrict__ out,
                           int D) {
    __shared__ float red[256];
    int tok = blockIdx.x, tid = threadIdx.x;
    float v = mo[(size_t)tok * D + tid];
    red[tid] = v * v;
    __syncthreads();
    for (int s = blockDim.x >> 1; s > 0; s >>= 1) {
        if (tid < s) red[tid] += red[tid + s];
        __syncthreads();
    }
    float rs = rsqrtf(red[0] / (float)D + 1e-5f);
    out[(size_t)tok * D + tid] = v * rs * w[tid] + (float)pre[(size_t)tok * D + tid];
}

// ---------------------------------------------------------------------------
// WMMA GEMM:  C[M,N] = act(A[M,K] * W + bias) + resid
// A: M x K fp16 row-major.  Wp: packed fragments (see k_pack_w).
// Block = 128 threads (4 waves).  Each wave: 16 rows x (JT*16) cols strip,
// JT accumulators, A fragment reused across the JT WMMAs per k-step.
// grid = (Ntiles/JT, M/64).  JT is compile-time -> no branches in k-loop.
// revA: read A rows sequence-reversed within each batch chunk of `seqlen`.
// ---------------------------------------------------------------------------
template <int JT>
__global__ void k_wmma_gemm(const _Float16* __restrict__ A,
                            const _Float16* __restrict__ Wp,
                            const float* __restrict__ bias,
                            const float* __restrict__ resid,
                            _Float16* __restrict__ out16,
                            float* __restrict__ out32,
                            int M, int N, int K, int act, int revA, int seqlen) {
    int lane = threadIdx.x & 31;
    int wave = threadIdx.x >> 5;
    int half = lane >> 4;        // 0: lanes 0-15, 1: lanes 16-31
    int l16  = lane & 15;
    int Ktiles = K >> 5;
    int nt0 = blockIdx.x * JT;
    int mt  = blockIdx.y * 4 + wave;

    int rowA = mt * 16 + l16;
    if (revA) {
        int b = rowA / seqlen, t = rowA % seqlen;
        rowA = b * seqlen + (seqlen - 1 - t);
    }
    const _Float16* Abase = A + (size_t)rowA * K + half * 8;
    const _Float16* Bbase = Wp + ((size_t)nt0 * Ktiles * 32 + lane) * 16;
    const size_t btile = (size_t)Ktiles * 512;   // halves per n-tile block

    v8f acc[JT];
#pragma unroll
    for (int j = 0; j < JT; ++j)
#pragma unroll
        for (int r = 0; r < 8; ++r) acc[j][r] = 0.0f;

    for (int kt = 0; kt < Ktiles; ++kt) {
        const v8h* Ap = (const v8h*)(Abase + kt * 32);
        v8h alo = Ap[0];        // K = kt*32 + half*8 + 0..7
        v8h ahi = Ap[2];        // K = kt*32 + half*8 + 16..23
        v16h a = __builtin_shufflevector(alo, ahi, 0, 1, 2, 3, 4, 5, 6, 7,
                                         8, 9, 10, 11, 12, 13, 14, 15);
        const _Float16* Bp = Bbase + (size_t)kt * 512;
#pragma unroll
        for (int j = 0; j < JT; ++j) {
            v16h b = *(const v16h*)(Bp + (size_t)j * btile);
            acc[j] = __builtin_amdgcn_wmma_f32_16x16x32_f16(
                false, a, false, b, (short)0, acc[j], false, false);
        }
    }

#pragma unroll
    for (int j = 0; j < JT; ++j) {
        int col = (nt0 + j) * 16 + l16;
        float bv = bias ? bias[col] : 0.0f;
#pragma unroll
        for (int r = 0; r < 8; ++r) {
            int row = mt * 16 + r + 8 * half;   // D layout: M = r + 8*half
            float v = acc[j][r] + bv;
            if (act == 1) v = silu_f(v);
            if (resid) v += resid[(size_t)row * N + col];
            if (out16) out16[(size_t)row * N + col] = (_Float16)v;
            if (out32) out32[(size_t)row * N + col] = v;
        }
    }
}

// ---------------------------------------------------------------------------
// Causal depthwise conv1d (K=4) + SiLU over xx = xz[:, :DINNER].
// ---------------------------------------------------------------------------
__global__ void k_conv_silu(const _Float16* __restrict__ xz16,
                            const float* __restrict__ conv_w,
                            const float* __restrict__ conv_b,
                            _Float16* __restrict__ xc16) {
    int idx = blockIdx.x * blockDim.x + threadIdx.x;
    if (idx >= TOKENS * DINNER) return;
    int d = idx % DINNER;
    int tok = idx / DINNER;
    int t = tok % SEQ, b = tok / SEQ;
    float acc = conv_b[d];
#pragma unroll
    for (int k = 0; k < 4; ++k) {
        int ts = t - 3 + k;
        if (ts >= 0)
            acc += conv_w[d * 4 + k] *
                   (float)xz16[((size_t)(b * SEQ + ts)) * (2 * DINNER) + d];
    }
    xc16[(size_t)tok * DINNER + d] = (_Float16)silu_f(acc);
}

// ---------------------------------------------------------------------------
// Selective scan: one lane per (batch, channel).  16 states in registers.
// dbl: tok x 48 fp32 = [dt_raw(16) | B(16) | C(16)].
// ---------------------------------------------------------------------------
__global__ void k_scan(const float* __restrict__ dbl,
                       const _Float16* __restrict__ xc16,
                       const float* __restrict__ dt_w,
                       const float* __restrict__ dt_b,
                       const float* __restrict__ A_log,
                       const float* __restrict__ Dp,
                       float* __restrict__ y32) {
    int gid = blockIdx.x * blockDim.x + threadIdx.x;
    if (gid >= NBATCH * DINNER) return;
    int b = gid / DINNER, d = gid % DINNER;

    float A[DSTATE], h[DSTATE], wdt[DSTATE];
#pragma unroll
    for (int n = 0; n < DSTATE; ++n) {
        A[n]   = -__expf(A_log[d * DSTATE + n]);
        wdt[n] = dt_w[d * DTRANK + n];
        h[n]   = 0.0f;
    }
    float bdt = dt_b[d], Dd = Dp[d];

    for (int t = 0; t < SEQ; ++t) {
        size_t tok = (size_t)b * SEQ + t;
        const float* row = dbl + tok * 48;
        float dt = bdt;
#pragma unroll
        for (int i = 0; i < DTRANK; ++i) dt += row[i] * wdt[i];
        dt = (dt > 20.0f) ? dt : log1pf(__expf(dt));   // softplus
        float xx = (float)xc16[tok * DINNER + d];
        float dtxx = dt * xx;
        float y = 0.0f;
#pragma unroll
        for (int n = 0; n < DSTATE; ++n) {
            float dA = __expf(dt * A[n]);
            h[n] = dA * h[n] + dtxx * row[16 + n];
            y += h[n] * row[32 + n];
        }
        y32[tok * DINNER + d] = y + xx * Dd;
    }
}

// gated16 = (y32) * silu(z),  z = xz[:, DINNER + d]
__global__ void k_gate(const float* __restrict__ y32,
                       const _Float16* __restrict__ xz16,
                       _Float16* __restrict__ gated16) {
    int idx = blockIdx.x * blockDim.x + threadIdx.x;
    if (idx >= TOKENS * DINNER) return;
    int d = idx % DINNER;
    int tok = idx / DINNER;
    float z = (float)xz16[(size_t)tok * (2 * DINNER) + DINNER + d];
    gated16[idx] = (_Float16)(y32[idx] * silu_f(z));
}

// fused16 = w32 * (fwd + bwd[seq-reversed])
__global__ void k_combine(const float* __restrict__ w32,
                          const float* __restrict__ ffin,
                          const float* __restrict__ bfin,
                          _Float16* __restrict__ fused16) {
    int idx = blockIdx.x * blockDim.x + threadIdx.x;
    if (idx >= TOKENS * DMODEL) return;
    int o = idx % DMODEL;
    int tok = idx / DMODEL;
    int t = tok % SEQ, b = tok / SEQ;
    size_t rtok = (size_t)b * SEQ + (SEQ - 1 - t);
    float v = ffin[idx] + bfin[rtok * DMODEL + o];
    fused16[idx] = (_Float16)(w32[idx] * v);
}

// ---------------------------------------------------------------------------
extern "C" void kernel_launch(void* const* d_in, const int* in_sizes, int n_in,
                              void* d_out, int out_size, void* d_ws, size_t ws_size,
                              hipStream_t stream) {
    (void)in_sizes; (void)n_in; (void)out_size; (void)ws_size;

    const float* x        = (const float*)d_in[0];
    const float* W_in     = (const float*)d_in[1];
    const float* b_in     = (const float*)d_in[2];
    const float* W_wp     = (const float*)d_in[3];
    const float* b_wp     = (const float*)d_in[4];
    const float* W_fp     = (const float*)d_in[5];
    const float* b_fp     = (const float*)d_in[6];
    const float* W_bp     = (const float*)d_in[7];
    const float* b_bp     = (const float*)d_in[8];
    const float* W_out    = (const float*)d_in[9];
    const float* b_out    = (const float*)d_in[10];
    const float* norm_top = (const float*)d_in[11];

    const float *in_w[2], *conv_w[2], *conv_b[2], *xproj_w[2], *dt_w[2],
                *dt_b[2], *A_log[2], *Dp[2], *out_w[2], *norm_w[2];
    for (int dir = 0; dir < 2; ++dir) {
        int base = 12 + 10 * dir;
        in_w[dir]    = (const float*)d_in[base + 0];
        conv_w[dir]  = (const float*)d_in[base + 1];
        conv_b[dir]  = (const float*)d_in[base + 2];
        xproj_w[dir] = (const float*)d_in[base + 3];
        dt_w[dir]    = (const float*)d_in[base + 4];
        dt_b[dir]    = (const float*)d_in[base + 5];
        A_log[dir]   = (const float*)d_in[base + 6];
        Dp[dir]      = (const float*)d_in[base + 7];
        out_w[dir]   = (const float*)d_in[base + 8];
        norm_w[dir]  = (const float*)d_in[base + 9];
    }

    // ---- workspace bump allocator --------------------------------------
    char* ws = (char*)d_ws;
    size_t off = 0;
    auto alloc = [&](size_t bytes) -> void* {
        void* p = ws + off;
        off += (bytes + 255) & ~(size_t)255;
        return p;
    };
    const size_t T = TOKENS;

    _Float16* xn16   = (_Float16*)alloc(T * INDIM  * 2);
    _Float16* ssm16  = (_Float16*)alloc(T * DMODEL * 2);
    float*    w32    = (float*)   alloc(T * DMODEL * 4);
    _Float16* pre16[2];
    float*    final32[2];
    for (int d = 0; d < 2; ++d) {
        pre16[d]   = (_Float16*)alloc(T * DMODEL * 2);
        final32[d] = (float*)   alloc(T * DMODEL * 4);
    }
    _Float16* fused16 = (_Float16*)alloc(T * DMODEL * 2);

    // packed weights (fragment order); size = N*K halves each
    _Float16* WinP  = (_Float16*)alloc((size_t)DMODEL * INDIM  * 2);
    _Float16* WwpP  = (_Float16*)alloc((size_t)DMODEL * INDIM  * 2);
    _Float16* WfpP  = (_Float16*)alloc((size_t)DMODEL * DMODEL * 2);
    _Float16* WbpP  = (_Float16*)alloc((size_t)DMODEL * DMODEL * 2);
    _Float16* WoutP = (_Float16*)alloc((size_t)INDIM  * DMODEL * 2);
    _Float16 *inwP[2], *xprojP[2], *outwP[2];
    for (int d = 0; d < 2; ++d) {
        inwP[d]   = (_Float16*)alloc((size_t)(2 * DINNER) * DMODEL * 2);
        xprojP[d] = (_Float16*)alloc((size_t)48 * DINNER * 2);
        outwP[d]  = (_Float16*)alloc((size_t)DMODEL * DINNER * 2);
    }
    // per-direction scratch (reused across dirs)
    _Float16* xz16    = (_Float16*)alloc(T * 2 * DINNER * 2);
    _Float16* xc16    = (_Float16*)alloc(T * DINNER * 2);
    float*    dbl32   = (float*)   alloc(T * 48 * 4);
    float*    y32     = (float*)   alloc(T * DINNER * 4);
    _Float16* gated16 = (_Float16*)alloc(T * DINNER * 2);
    float*    mout32  = (float*)   alloc(T * DMODEL * 4);

    // ---- weight packing --------------------------------------------------
    auto pack = [&](const float* W, _Float16* Wp, int N, int K) {
        int n = N * K;
        k_pack_w<<<(n + 255) / 256, 256, 0, stream>>>(W, Wp, N, K);
    };
    pack(W_in,  WinP,  DMODEL, INDIM);
    pack(W_wp,  WwpP,  DMODEL, INDIM);
    pack(W_fp,  WfpP,  DMODEL, DMODEL);
    pack(W_bp,  WbpP,  DMODEL, DMODEL);
    pack(W_out, WoutP, INDIM,  DMODEL);
    for (int d = 0; d < 2; ++d) {
        pack(in_w[d],    inwP[d],   2 * DINNER, DMODEL);
        pack(xproj_w[d], xprojP[d], 48,         DINNER);
        pack(out_w[d],   outwP[d],  DMODEL,     DINNER);
    }

    // N multiple of 64 -> 4 tiles/wave; N==48 -> 3 tiles/wave.
    auto gemm = [&](const _Float16* A, const _Float16* Wp, const float* bias,
                    const float* resid, _Float16* o16, float* o32,
                    int M, int N, int K, int act, int rev) {
        int Ntiles = N / 16;
        if (Ntiles % 4 == 0) {
            dim3 grid(Ntiles / 4, M / 64);
            k_wmma_gemm<4><<<grid, 128, 0, stream>>>(A, Wp, bias, resid, o16,
                                                     o32, M, N, K, act, rev, SEQ);
        } else {
            dim3 grid(Ntiles / 3, M / 64);
            k_wmma_gemm<3><<<grid, 128, 0, stream>>>(A, Wp, bias, resid, o16,
                                                     o32, M, N, K, act, rev, SEQ);
        }
    };

    // ---- pipeline -------------------------------------------------------
    // top rmsnorm -> xn16
    k_rmsnorm16<<<TOKENS, INDIM, 0, stream>>>(x, norm_top, xn16, INDIM);
    // ssm_in = xn @ W_in^T + b_in
    gemm(xn16, WinP, b_in, nullptr, ssm16, nullptr, TOKENS, DMODEL, INDIM, 0, 0);
    // w = silu(xn @ W_wp^T + b_wp)
    gemm(xn16, WwpP, b_wp, nullptr, nullptr, w32, TOKENS, DMODEL, INDIM, 1, 0);
    // pre_f = ssm_in @ W_fp^T + b_fp ; pre_bk = reversed(ssm_in) @ W_bp^T + b_bp
    gemm(ssm16, WfpP, b_fp, nullptr, pre16[0], nullptr, TOKENS, DMODEL, DMODEL, 0, 0);
    gemm(ssm16, WbpP, b_bp, nullptr, pre16[1], nullptr, TOKENS, DMODEL, DMODEL, 0, 1);

    for (int dir = 0; dir < 2; ++dir) {
        // xz = pre @ in_w^T
        gemm(pre16[dir], inwP[dir], nullptr, nullptr, xz16, nullptr,
             TOKENS, 2 * DINNER, DMODEL, 0, 0);
        // depthwise causal conv + silu
        {
            int n = TOKENS * DINNER;
            k_conv_silu<<<(n + 255) / 256, 256, 0, stream>>>(xz16, conv_w[dir],
                                                             conv_b[dir], xc16);
        }
        // dbl = xc @ xproj^T   (48 = dt_rank 16 + B 16 + C 16)
        gemm(xc16, xprojP[dir], nullptr, nullptr, nullptr, dbl32,
             TOKENS, 48, DINNER, 0, 0);
        // selective scan
        {
            int n = NBATCH * DINNER;
            k_scan<<<(n + 255) / 256, 256, 0, stream>>>(dbl32, xc16, dt_w[dir],
                                                        dt_b[dir], A_log[dir],
                                                        Dp[dir], y32);
        }
        // gate with silu(z)
        {
            int n = TOKENS * DINNER;
            k_gate<<<(n + 255) / 256, 256, 0, stream>>>(y32, xz16, gated16);
        }
        // mamba out proj
        gemm(gated16, outwP[dir], nullptr, nullptr, nullptr, mout32,
             TOKENS, DMODEL, DINNER, 0, 0);
        // post-norm + residual
        k_postnorm<<<TOKENS, DMODEL, 0, stream>>>(mout32, norm_w[dir],
                                                  pre16[dir], final32[dir],
                                                  DMODEL);
    }

    // fused = w * (fwd + reversed(bwd))
    {
        int n = TOKENS * DMODEL;
        k_combine<<<(n + 255) / 256, 256, 0, stream>>>(w32, final32[0],
                                                       final32[1], fused16);
    }
    // out = fused @ W_out^T + b_out + x
    gemm(fused16, WoutP, b_out, x, nullptr, (float*)d_out,
         TOKENS, INDIM, DMODEL, 0, 0);
}
